// Transformer_39032662786309
// MI455X (gfx1250) — compile-verified, matrix operand order checked by reading, not run on hardware
//
#include <hip/hip_runtime.h>

// ---------------------------------------------------------------------------
// Transformer forward (enc-dec, 6+6 layers) for MI455X / gfx1250, wave32.
// - All GEMM-shaped math on v_wmma_f32_16x16x32_bf16 (bf16 in, fp32 acc).
// - All WMMA operands staged K-contiguous in LDS so every fragment is two
//   aligned ds_load_b128: weights pre-transposed to [N,K] bf16, V written
//   [b,h,dk,seq] from its projection epilogue.
// - Global->LDS staging via GLOBAL_LOAD_ASYNC_TO_LDS_B128 (ASYNCcnt) with
//   double-buffered LDS tiles overlapping loads with the WMMA pipe.
// - GEMM/score kernels: 128x128 block, 8 waves, 32x64 per wave =
//   8 WMMAs per k-step vs 6 fragment loads (LDS-read pressure halved).
// ---------------------------------------------------------------------------

#define D_MODEL 512
#define N_HEADS 8
#define D_KH    64
#define N_LAYERS 6
#define D_FF    2048
#define VOCAB   32000
#define BATCH   8
#define SEQ     512
#define NTOK    (BATCH * SEQ)

typedef __bf16 bf16;
typedef __attribute__((ext_vector_type(16))) __bf16 v16bf;
typedef __attribute__((ext_vector_type(8)))  __bf16 v8bf;
typedef __attribute__((ext_vector_type(8)))  float  v8f;

// ------------------------------------------------------------ async staging
__device__ __forceinline__ void async_cp16(bf16* lds, const bf16* g) {
  unsigned l = (unsigned)(unsigned long long)lds;  // LDS offset = addr[31:0]
  asm volatile("global_load_async_to_lds_b128 %0, %1, off"
               :: "v"(l), "v"(g) : "memory");
}
__device__ __forceinline__ void wait_async() {
  asm volatile("s_wait_asynccnt 0" ::: "memory");
}

// ---------------------------------------------------------------- utilities

// fused bf16 convert + transpose: in [K,N] f32 -> out [N,K] bf16
__global__ void cvt_transpose_kernel(const float* __restrict__ in,
                                     bf16* __restrict__ out, int K, int N) {
  __shared__ float t[32][33];
  int kb = blockIdx.y * 32, nb = blockIdx.x * 32;
  int tx = threadIdx.x & 31, ty = threadIdx.x >> 5;  // 256 thr: ty 0..7
#pragma unroll
  for (int j = 0; j < 32; j += 8)
    t[ty + j][tx] = in[(size_t)(kb + ty + j) * N + nb + tx];
  __syncthreads();
#pragma unroll
  for (int j = 0; j < 32; j += 8)
    out[(size_t)(nb + ty + j) * K + kb + tx] = (bf16)t[tx][ty + j];
}

// token embedding * sqrt(d_model) + sinusoidal PE, fp32 + bf16 outputs
__global__ void embed_kernel(const int* __restrict__ tok,
                             const float* __restrict__ emb,
                             float* __restrict__ xf, bf16* __restrict__ xb) {
  int t = blockIdx.x;            // token index (b*SEQ + pos)
  int pos = t & (SEQ - 1);
  int id = tok[t];
  const float scale = 22.62741699796952f;  // sqrt(512)
  for (int d = threadIdx.x; d < D_MODEL; d += blockDim.x) {
    float freq = __expf(-0.017988946039016f * (float)(d & ~1));
    float ang = (float)pos * freq;
    float pe = (d & 1) ? __cosf(ang) : __sinf(ang);
    float v = emb[(size_t)id * D_MODEL + d] * scale + pe;
    xf[(size_t)t * D_MODEL + d] = v;
    xb[(size_t)t * D_MODEL + d] = (bf16)v;
  }
}

// y = LayerNorm(resid + delta); wave-per-row (512 elems, 16/lane, wave32)
__global__ void add_ln_kernel(const float* __restrict__ resid,
                              const float* __restrict__ delta,
                              const float* __restrict__ g,
                              const float* __restrict__ b,
                              float* __restrict__ yf, bf16* __restrict__ yb) {
  int row = blockIdx.x * 8 + (threadIdx.x >> 5);
  int lane = threadIdx.x & 31;
  const float* xr = resid + (size_t)row * D_MODEL;
  const float* xd = delta + (size_t)row * D_MODEL;
  float v[16], s = 0.f, s2 = 0.f;
#pragma unroll
  for (int j = 0; j < 16; j++) {
    float t = xr[lane + 32 * j] + xd[lane + 32 * j];
    v[j] = t; s += t; s2 += t * t;
  }
#pragma unroll
  for (int o = 16; o; o >>= 1) {
    s  += __shfl_xor(s,  o, 32);
    s2 += __shfl_xor(s2, o, 32);
  }
  float mean = s * (1.f / D_MODEL);
  float var = s2 * (1.f / D_MODEL) - mean * mean;
  float rstd = rsqrtf(var + 1e-5f);
#pragma unroll
  for (int j = 0; j < 16; j++) {
    int d = lane + 32 * j;
    float t = (v[j] - mean) * rstd * g[d] + b[d];
    yf[(size_t)row * D_MODEL + d] = t;
    yb[(size_t)row * D_MODEL + d] = (bf16)t;
  }
}

// wave-per-row softmax over SEQ keys, fp32 scores -> bf16 probabilities
__global__ void softmax_kernel(const float* __restrict__ S,
                               bf16* __restrict__ P) {
  size_t row = (size_t)blockIdx.x * 8 + (threadIdx.x >> 5);
  int lane = threadIdx.x & 31;
  const float* r = S + row * SEQ;
  float v[16], mx = -1e30f;
#pragma unroll
  for (int j = 0; j < 16; j++) { v[j] = r[lane + 32 * j]; mx = fmaxf(mx, v[j]); }
#pragma unroll
  for (int o = 16; o; o >>= 1) mx = fmaxf(mx, __shfl_xor(mx, o, 32));
  float s = 0.f;
#pragma unroll
  for (int j = 0; j < 16; j++) { v[j] = __expf(v[j] - mx); s += v[j]; }
#pragma unroll
  for (int o = 16; o; o >>= 1) s += __shfl_xor(s, o, 32);
  float inv = 1.f / s;
  bf16* p = P + row * SEQ;
#pragma unroll
  for (int j = 0; j < 16; j++) p[lane + 32 * j] = (bf16)(v[j] * inv);
}

// ------------------------------------------------- WMMA fragment helper
// bf16 16x32 operand fragment (ISA 7.12.2): lanes 0-15 -> row=lane,
// K {0-7,16-23}; lanes 16-31 -> row=lane-16, K {8-15,24-31}.
// `row` points at a K-contiguous 32-element LDS row; kb = 8*(lane>>4).
// Lowers to two aligned ds_load_b128.
__device__ __forceinline__ v16bf frag16(const bf16* row, int kb) {
  v8bf lo = *(const v8bf*)(row + kb);
  v8bf hi = *(const v8bf*)(row + kb + 16);
  v16bf a;
#pragma unroll
  for (int j = 0; j < 8; j++) { a[j] = lo[j]; a[j + 8] = hi[j]; }
  return a;
}

#define BM 128
#define BN 128          // gemm / score block N
#define BK 32
#define LDS_K (BK + 8)  // padded K-contiguous LDS row (80B, 16B-aligned)

// --------------------------------------------- generic bf16 WMMA GEMM
// C[M,N] = A[M,K] @ Bt[N,K]ᵀ (+bias, optional ReLU). Both operands staged
// K-contiguous. 128x128 block, 8 waves (4x2), each wave 32x64 = 2x4 WMMA.
// Outputs: Cf fp32 and/or Cb bf16 (row-major, ldc) and/or CbT (V layout
// [b,h,dk,seq]). Requires M%128==0, N%128==0, K%32==0.
__global__ __launch_bounds__(256) void gemm_bf16_kernel(
    const bf16* __restrict__ A, int lda, const bf16* __restrict__ Bt, int ldb,
    const float* __restrict__ bias, float* __restrict__ Cf,
    bf16* __restrict__ Cb, bf16* __restrict__ CbT, int ldc, int M, int N,
    int K, int relu) {
  __shared__ bf16 As[2][BM][LDS_K];
  __shared__ bf16 Bs[2][BN][LDS_K];
  const int tid = threadIdx.x, lane = tid & 31;
  const int wave = tid >> 5, wrow = wave >> 1, wcol = wave & 1;
  const int l16 = lane & 15, lhi = lane >> 4, kb = lhi * 8;
  const int bm = blockIdx.y * BM, bn = blockIdx.x * BN;
  const int r = tid >> 1, c = (tid & 1) * 16;  // 128 rows x 2x16B chunks
  auto stage = [&](int buf, int k0) {
    async_cp16(&As[buf][r][c],     A  + (size_t)(bm + r) * lda + k0 + c);
    async_cp16(&As[buf][r][c + 8], A  + (size_t)(bm + r) * lda + k0 + c + 8);
    async_cp16(&Bs[buf][r][c],     Bt + (size_t)(bn + r) * ldb + k0 + c);
    async_cp16(&Bs[buf][r][c + 8], Bt + (size_t)(bn + r) * ldb + k0 + c + 8);
  };
  v8f acc[2][4] = {};
  stage(0, 0);
  wait_async();
  __syncthreads();
  const int nk = K / BK;
  for (int i = 0; i < nk; i++) {
    int buf = i & 1;
    if (i + 1 < nk) stage(buf ^ 1, (i + 1) * BK);
    v16bf af[2], bf_[4];
#pragma unroll
    for (int fm = 0; fm < 2; fm++)
      af[fm] = frag16(&As[buf][wrow * 32 + fm * 16 + l16][0], kb);
#pragma unroll
    for (int fn = 0; fn < 4; fn++)
      bf_[fn] = frag16(&Bs[buf][wcol * 64 + fn * 16 + l16][0], kb);
#pragma unroll
    for (int fm = 0; fm < 2; fm++)
#pragma unroll
      for (int fn = 0; fn < 4; fn++)
        acc[fm][fn] = __builtin_amdgcn_wmma_f32_16x16x32_bf16(
            false, af[fm], false, bf_[fn], (short)0, acc[fm][fn], false, false);
    if (i + 1 < nk) {
      wait_async();
      __syncthreads();
    }
  }
#pragma unroll
  for (int fm = 0; fm < 2; fm++)
#pragma unroll
    for (int fn = 0; fn < 4; fn++) {
      int n = bn + wcol * 64 + fn * 16 + l16;
      float bv = bias ? bias[n] : 0.f;
#pragma unroll
      for (int rr = 0; rr < 8; rr++) {
        int m = bm + wrow * 32 + fm * 16 + rr + 8 * lhi;
        float v = acc[fm][fn][rr] + bv;
        if (relu) v = fmaxf(v, 0.f);
        if (Cf) Cf[(size_t)m * ldc + n] = v;
        if (Cb) Cb[(size_t)m * ldc + n] = (bf16)v;
        if (CbT) {  // V projection: [tok, h*64+dk] -> [b, h, dk, seq]
          int bq = m >> 9, s = m & (SEQ - 1);
          int hh = n >> 6, dk = n & (D_KH - 1);
          CbT[(((size_t)bq * N_HEADS + hh) * D_KH + dk) * SEQ + s] = (bf16)v;
        }
      }
    }
}

// ------------------------------ attention scores: S = Q Kᵀ / 8, masked
// per (b,h): A = Q_bh [Sq,64]; Kᵀ operand = K_bh [Sk,64] (K-contiguous).
// 128x128 tile of the score matrix per block; fused scale + mask.
__global__ __launch_bounds__(256) void attn_score_kernel(
    const bf16* __restrict__ Q, const bf16* __restrict__ Kmat,
    float* __restrict__ S, const int* __restrict__ tok, int causal) {
  __shared__ bf16 As[2][BM][LDS_K];
  __shared__ bf16 Bs[2][BN][LDS_K];
  int z = blockIdx.z, b = z >> 3, h = z & 7;
  const bf16* Qb = Q + (size_t)b * SEQ * D_MODEL + h * D_KH;
  const bf16* Kb = Kmat + (size_t)b * SEQ * D_MODEL + h * D_KH;
  const int tid = threadIdx.x, lane = tid & 31;
  const int wave = tid >> 5, wrow = wave >> 1, wcol = wave & 1;
  const int l16 = lane & 15, lhi = lane >> 4, kb = lhi * 8;
  const int bm = blockIdx.y * BM, bn = blockIdx.x * BN;
  const int r = tid >> 1, c = (tid & 1) * 16;
  auto stage = [&](int buf, int k0) {
    async_cp16(&As[buf][r][c],     Qb + (size_t)(bm + r) * D_MODEL + k0 + c);
    async_cp16(&As[buf][r][c + 8], Qb + (size_t)(bm + r) * D_MODEL + k0 + c + 8);
    async_cp16(&Bs[buf][r][c],     Kb + (size_t)(bn + r) * D_MODEL + k0 + c);
    async_cp16(&Bs[buf][r][c + 8], Kb + (size_t)(bn + r) * D_MODEL + k0 + c + 8);
  };
  v8f acc[2][4] = {};
  stage(0, 0);
  wait_async();
  __syncthreads();
  const int nk = D_KH / BK;  // 2
  for (int i = 0; i < nk; i++) {
    int buf = i & 1;
    if (i + 1 < nk) stage(buf ^ 1, (i + 1) * BK);
    v16bf af[2], bf_[4];
#pragma unroll
    for (int fm = 0; fm < 2; fm++)
      af[fm] = frag16(&As[buf][wrow * 32 + fm * 16 + l16][0], kb);
#pragma unroll
    for (int fn = 0; fn < 4; fn++)
      bf_[fn] = frag16(&Bs[buf][wcol * 64 + fn * 16 + l16][0], kb);
#pragma unroll
    for (int fm = 0; fm < 2; fm++)
#pragma unroll
      for (int fn = 0; fn < 4; fn++)
        acc[fm][fn] = __builtin_amdgcn_wmma_f32_16x16x32_bf16(
            false, af[fm], false, bf_[fn], (short)0, acc[fm][fn], false, false);
    if (i + 1 < nk) {
      wait_async();
      __syncthreads();
    }
  }
  const int* kt = tok + b * SEQ;
  float* Sz = S + (size_t)z * SEQ * SEQ;
#pragma unroll
  for (int fm = 0; fm < 2; fm++)
#pragma unroll
    for (int fn = 0; fn < 4; fn++) {
      int n = bn + wcol * 64 + fn * 16 + l16;
      int pad_ok = (kt[n] != 0);
#pragma unroll
      for (int rr = 0; rr < 8; rr++) {
        int m = bm + wrow * 32 + fm * 16 + rr + 8 * lhi;
        bool valid = pad_ok && (!causal || n <= m);
        Sz[(size_t)m * SEQ + n] = valid ? acc[fm][fn][rr] * 0.125f : -1e9f;
      }
    }
}

// ------------------------------ attention output: O_bh = P_bh @ V_bh
// A = P [Sq,Sk] row-major (K-contiguous); B operand = Vt[b,h,dk,seq]
// (K-contiguous along seq). N = 64 -> 128x64 block, wave 32x32 (2x2 WMMA).
__global__ __launch_bounds__(256) void attn_out_kernel(
    const bf16* __restrict__ P, const bf16* __restrict__ Vt,
    bf16* __restrict__ O) {
  __shared__ bf16 As[2][BM][LDS_K];
  __shared__ bf16 Bs[2][64][LDS_K];
  int z = blockIdx.z, b = z >> 3, h = z & 7;
  const bf16* Pb = P + (size_t)z * SEQ * SEQ;
  const bf16* Vb = Vt + (size_t)z * D_KH * SEQ;  // [dk][seq]
  const int tid = threadIdx.x, lane = tid & 31;
  const int wave = tid >> 5, wrow = wave >> 1, wcol = wave & 1;
  const int l16 = lane & 15, lhi = lane >> 4, kb = lhi * 8;
  const int bm = blockIdx.y * BM;  // bn == 0 (N = 64 exactly)
  const int r = tid >> 2, c = (tid & 3) * 8;  // 64 rows x 4 chunks of 8
  auto stage = [&](int buf, int k0) {
    async_cp16(&As[buf][r][c],      Pb + (size_t)(bm + r)      * SEQ + k0 + c);
    async_cp16(&As[buf][r + 64][c], Pb + (size_t)(bm + r + 64) * SEQ + k0 + c);
    async_cp16(&Bs[buf][r][c],      Vb + (size_t)r * SEQ + k0 + c);
  };
  v8f acc[2][2] = {};
  stage(0, 0);
  wait_async();
  __syncthreads();
  const int nk = SEQ / BK;  // 16
  for (int i = 0; i < nk; i++) {
    int buf = i & 1;
    if (i + 1 < nk) stage(buf ^ 1, (i + 1) * BK);
    v16bf af[2], bf_[2];
#pragma unroll
    for (int fm = 0; fm < 2; fm++)
      af[fm] = frag16(&As[buf][wrow * 32 + fm * 16 + l16][0], kb);
#pragma unroll
    for (int fn = 0; fn < 2; fn++)
      bf_[fn] = frag16(&Bs[buf][wcol * 32 + fn * 16 + l16][0], kb);
#pragma unroll
    for (int fm = 0; fm < 2; fm++)
#pragma unroll
      for (int fn = 0; fn < 2; fn++)
        acc[fm][fn] = __builtin_amdgcn_wmma_f32_16x16x32_bf16(
            false, af[fm], false, bf_[fn], (short)0, acc[fm][fn], false, false);
    if (i + 1 < nk) {
      wait_async();
      __syncthreads();
    }
  }
#pragma unroll
  for (int fm = 0; fm < 2; fm++)
#pragma unroll
    for (int fn = 0; fn < 2; fn++) {
      int n = wcol * 32 + fn * 16 + l16;
#pragma unroll
      for (int rr = 0; rr < 8; rr++) {
        int m = bm + wrow * 32 + fm * 16 + rr + 8 * lhi;
        O[((size_t)b * SEQ + m) * D_MODEL + h * D_KH + n] =
            (bf16)acc[fm][fn][rr];
      }
    }
}

// --------------------------------------------------------------- host side

extern "C" void kernel_launch(void* const* d_in, const int* in_sizes, int n_in,
                              void* d_out, int out_size, void* d_ws,
                              size_t ws_size, hipStream_t stream) {
  (void)in_sizes; (void)n_in; (void)out_size; (void)ws_size;
  // JAX pytree flatten order (dict keys sorted alphabetically at every level):
  // params.dec: b1n,b2n,b3n, cross{bk,bo,bq,bv,wk,wo,wq,wv}, ff{b1,b2,w1,w2},
  //             g1,g2,g3, self{bk,bo,bq,bv,wk,wo,wq,wv}
  // params.enc: attn{bk,bo,bq,bv,wk,wo,wq,wv}, b1n,b2n, ff{b1,b2,w1,w2}, g1,g2
  // out_b, out_w, src_emb, tgt_emb, then src, tgt.
  auto f = [&](int i) { return (const float*)d_in[i]; };
  const float *d_b1n = f(0), *d_b2n = f(1), *d_b3n = f(2);
  const float *dc_bk = f(3), *dc_bo = f(4), *dc_bq = f(5), *dc_bv = f(6);
  const float *dc_wk = f(7), *dc_wo = f(8), *dc_wq = f(9), *dc_wv = f(10);
  const float *d_fb1 = f(11), *d_fb2 = f(12), *d_fw1 = f(13), *d_fw2 = f(14);
  const float *d_g1 = f(15), *d_g2 = f(16), *d_g3 = f(17);
  const float *ds_bk = f(18), *ds_bo = f(19), *ds_bq = f(20), *ds_bv = f(21);
  const float *ds_wk = f(22), *ds_wo = f(23), *ds_wq = f(24), *ds_wv = f(25);
  const float *ea_bk = f(26), *ea_bo = f(27), *ea_bq = f(28), *ea_bv = f(29);
  const float *ea_wk = f(30), *ea_wo = f(31), *ea_wq = f(32), *ea_wv = f(33);
  const float *e_b1n = f(34), *e_b2n = f(35);
  const float *e_fb1 = f(36), *e_fb2 = f(37), *e_fw1 = f(38), *e_fw2 = f(39);
  const float *e_g1 = f(40), *e_g2 = f(41);
  const float *out_b = f(42), *out_w = f(43);
  const float *src_emb = f(44), *tgt_emb = f(45);
  const int *src = (const int*)d_in[46], *tgt = (const int*)d_in[47];

  char* cur = (char*)d_ws;
  auto alloc = [&](size_t bytes) -> void* {
    void* p = cur; cur += (bytes + 255) & ~(size_t)255; return p;
  };
  // convert + transpose a [K,N] f32 weight into [N,K] bf16
  auto convT = [&](const float* s, int K, int N) -> bf16* {
    bf16* d = (bf16*)alloc((size_t)K * N * sizeof(bf16));
    cvt_transpose_kernel<<<dim3(N / 32, K / 32), 256, 0, stream>>>(s, d, K, N);
    return d;
  };

  const size_t W2 = (size_t)D_MODEL * D_MODEL;
  const size_t WF1 = (size_t)D_MODEL * D_FF, WF2 = (size_t)D_FF * D_MODEL;

  // --- convert weights to transposed bf16 (deterministic, every call) ---
  bf16 *eWQ[N_LAYERS], *eWK[N_LAYERS], *eWV[N_LAYERS], *eWO[N_LAYERS];
  bf16 *eW1[N_LAYERS], *eW2[N_LAYERS];
  bf16 *sWQ[N_LAYERS], *sWK[N_LAYERS], *sWV[N_LAYERS], *sWO[N_LAYERS];
  bf16 *cWQ[N_LAYERS], *cWK[N_LAYERS], *cWV[N_LAYERS], *cWO[N_LAYERS];
  bf16 *dW1[N_LAYERS], *dW2[N_LAYERS];
  for (int l = 0; l < N_LAYERS; l++) {
    eWQ[l] = convT(ea_wq + l * W2, D_MODEL, D_MODEL);
    eWK[l] = convT(ea_wk + l * W2, D_MODEL, D_MODEL);
    eWV[l] = convT(ea_wv + l * W2, D_MODEL, D_MODEL);
    eWO[l] = convT(ea_wo + l * W2, D_MODEL, D_MODEL);
    eW1[l] = convT(e_fw1 + l * WF1, D_MODEL, D_FF);
    eW2[l] = convT(e_fw2 + l * WF2, D_FF, D_MODEL);
    sWQ[l] = convT(ds_wq + l * W2, D_MODEL, D_MODEL);
    sWK[l] = convT(ds_wk + l * W2, D_MODEL, D_MODEL);
    sWV[l] = convT(ds_wv + l * W2, D_MODEL, D_MODEL);
    sWO[l] = convT(ds_wo + l * W2, D_MODEL, D_MODEL);
    cWQ[l] = convT(dc_wq + l * W2, D_MODEL, D_MODEL);
    cWK[l] = convT(dc_wk + l * W2, D_MODEL, D_MODEL);
    cWV[l] = convT(dc_wv + l * W2, D_MODEL, D_MODEL);
    cWO[l] = convT(dc_wo + l * W2, D_MODEL, D_MODEL);
    dW1[l] = convT(d_fw1 + l * WF1, D_MODEL, D_FF);
    dW2[l] = convT(d_fw2 + l * WF2, D_FF, D_MODEL);
  }
  bf16* outw = convT(out_w, D_MODEL, VOCAB);  // [32000, 512]

  // --- activation buffers ---
  float* xe_f = (float*)alloc((size_t)NTOK * D_MODEL * 4);
  bf16*  xe_b = (bf16*)alloc((size_t)NTOK * D_MODEL * 2);
  float* xd_f = (float*)alloc((size_t)NTOK * D_MODEL * 4);
  bf16*  xd_b = (bf16*)alloc((size_t)NTOK * D_MODEL * 2);
  bf16*  Qb = (bf16*)alloc((size_t)NTOK * D_MODEL * 2);
  bf16*  Kb = (bf16*)alloc((size_t)NTOK * D_MODEL * 2);
  bf16*  Vt = (bf16*)alloc((size_t)NTOK * D_MODEL * 2);  // [b,h,dk,seq]
  float* scores = (float*)alloc((size_t)BATCH * N_HEADS * SEQ * SEQ * 4);
  bf16*  probs  = (bf16*)alloc((size_t)BATCH * N_HEADS * SEQ * SEQ * 2);
  bf16*  attnO = (bf16*)alloc((size_t)NTOK * D_MODEL * 2);
  float* delta = (float*)alloc((size_t)NTOK * D_MODEL * 4);
  bf16*  hbf = (bf16*)alloc((size_t)NTOK * D_FF * 2);

  auto gemm = [&](const bf16* A, int lda, const bf16* Bt, int ldb,
                  const float* bias, float* Cf, bf16* Cb, bf16* CbT, int ldc,
                  int M, int N, int K, int relu) {
    gemm_bf16_kernel<<<dim3(N / BN, M / BM), 256, 0, stream>>>(
        A, lda, Bt, ldb, bias, Cf, Cb, CbT, ldc, M, N, K, relu);
  };
  auto ln = [&](float* xf, bf16* xb, const float* dlt, const float* g,
                const float* b) {
    add_ln_kernel<<<NTOK / 8, 256, 0, stream>>>(xf, dlt, g, b, xf, xb);
  };
  auto attn = [&](const bf16* xq, const bf16* xkv, bf16* Wq, bf16* Wk,
                  bf16* Wv, bf16* Wo, const float* bq, const float* bk,
                  const float* bv, const float* bo, const int* keytok,
                  int causal) {
    gemm(xq, D_MODEL, Wq, D_MODEL, bq, nullptr, Qb, nullptr, D_MODEL, NTOK,
         D_MODEL, D_MODEL, 0);
    gemm(xkv, D_MODEL, Wk, D_MODEL, bk, nullptr, Kb, nullptr, D_MODEL, NTOK,
         D_MODEL, D_MODEL, 0);
    gemm(xkv, D_MODEL, Wv, D_MODEL, bv, nullptr, nullptr, Vt, D_MODEL, NTOK,
         D_MODEL, D_MODEL, 0);
    attn_score_kernel<<<dim3(SEQ / BN, SEQ / BM, BATCH * N_HEADS), 256, 0,
                        stream>>>(Qb, Kb, scores, keytok, causal);
    softmax_kernel<<<(BATCH * N_HEADS * SEQ) / 8, 256, 0, stream>>>(scores,
                                                                    probs);
    attn_out_kernel<<<dim3(1, SEQ / BM, BATCH * N_HEADS), 256, 0, stream>>>(
        probs, Vt, attnO);
    gemm(attnO, D_MODEL, Wo, D_MODEL, bo, delta, nullptr, nullptr, D_MODEL,
         NTOK, D_MODEL, D_MODEL, 0);
  };
  auto ffn = [&](const bf16* xb, bf16* W1, bf16* W2, const float* b1,
                 const float* b2) {
    gemm(xb, D_MODEL, W1, D_MODEL, b1, nullptr, hbf, nullptr, D_FF, NTOK,
         D_FF, D_MODEL, 1);
    gemm(hbf, D_FF, W2, D_FF, b2, delta, nullptr, nullptr, D_MODEL, NTOK,
         D_MODEL, D_FF, 0);
  };

  // ---------------- encoder ----------------
  embed_kernel<<<NTOK, 128, 0, stream>>>(src, src_emb, xe_f, xe_b);
  for (int l = 0; l < N_LAYERS; l++) {
    attn(xe_b, xe_b, eWQ[l], eWK[l], eWV[l], eWO[l], ea_bq + l * D_MODEL,
         ea_bk + l * D_MODEL, ea_bv + l * D_MODEL, ea_bo + l * D_MODEL, src, 0);
    ln(xe_f, xe_b, delta, e_g1 + l * D_MODEL, e_b1n + l * D_MODEL);
    ffn(xe_b, eW1[l], eW2[l], e_fb1 + l * D_FF, e_fb2 + l * D_MODEL);
    ln(xe_f, xe_b, delta, e_g2 + l * D_MODEL, e_b2n + l * D_MODEL);
  }

  // ---------------- decoder ----------------
  embed_kernel<<<NTOK, 128, 0, stream>>>(tgt, tgt_emb, xd_f, xd_b);
  for (int l = 0; l < N_LAYERS; l++) {
    attn(xd_b, xd_b, sWQ[l], sWK[l], sWV[l], sWO[l], ds_bq + l * D_MODEL,
         ds_bk + l * D_MODEL, ds_bv + l * D_MODEL, ds_bo + l * D_MODEL, tgt, 1);
    ln(xd_f, xd_b, delta, d_g1 + l * D_MODEL, d_b1n + l * D_MODEL);
    attn(xd_b, xe_b, cWQ[l], cWK[l], cWV[l], cWO[l], dc_bq + l * D_MODEL,
         dc_bk + l * D_MODEL, dc_bv + l * D_MODEL, dc_bo + l * D_MODEL, src, 0);
    ln(xd_f, xd_b, delta, d_g2 + l * D_MODEL, d_b2n + l * D_MODEL);
    ffn(xd_b, dW1[l], dW2[l], d_fb1 + l * D_FF, d_fb2 + l * D_MODEL);
    ln(xd_f, xd_b, delta, d_g3 + l * D_MODEL, d_b3n + l * D_MODEL);
  }

  // final vocab projection: [4096,512] @ [512,32000] + out_b -> d_out (fp32)
  gemm(xd_b, D_MODEL, outw, D_MODEL, out_b, (float*)d_out, nullptr, nullptr,
       VOCAB, NTOK, VOCAB, D_MODEL, 0);
}
